// frac_fftformer_54039278518451
// MI455X (gfx1250) — matrix operand order repaired and checked by reading, not compiled, not run on hardware
//
#include <hip/hip_runtime.h>

// ---------------------------------------------------------------------------
// Deformable Conv v2 on MI455X (gfx1250), wave32 + WMMA bf16 + TDM.
//
// Sizes: B=2, C=128, H=W=128, N=9 (3x3), outc=128.
// Pipeline:
//   1) x NCHW -> NHWC f32 (LDS-tiled transpose, both sides coalesced)
//   2) repack w_p/w_m -> wA bf16 [32 x 1152]  (K = ij*128 + c, rows 27..31 = 0)
//      repack w_conv  -> wC bf16 [128 x 1152] (K = ij*128 + c, matches G)
//   3) GEMM-A (WMMA bf16): offs[b][27][pix] = wA x im2col(x) + bias
//      (K-order ij*128+c makes the im2col shift uniform per 32-wide K-step)
//   4) zero G, then sampling: bilinear gather (channel-invariant weights,
//      branch-free via weight masking), sigmoid modulation, scatter to
//      G[b][ij][pix][c] bf16 (contiguous 256B per wave)
//   5) GEMM-C (WMMA bf16): out[b][oc][pix] = wC x G, A/B tiles staged by the
//      Tensor Data Mover (tensor_load_to_lds, double-buffered, overlap DMA
//      with WMMA), s_wait_tensorcnt for completion.
//
// Workspace (bytes): xn f32 @0 (16,777,216) | offs f32 @16,777,216 (3,538,944)
//   wA bf16 @20,316,160 (73,728) | wC bf16 @20,389,888 (294,912)
//   G bf16 [2][9][16384][128] @20,684,800 (75,497,472)   total ~96.2 MB
// ---------------------------------------------------------------------------

typedef __attribute__((ext_vector_type(16))) __bf16 v16bf;
typedef __attribute__((ext_vector_type(8)))  float  v8f;

#define CC   128
#define HH   128
#define WW   128
#define HW   16384
#define KTOT 1152      // 9*128
#define KSTEPS 36      // 1152/32

#if defined(__has_builtin)
#  if __has_builtin(__builtin_amdgcn_tensor_load_to_lds) && \
      __has_builtin(__builtin_amdgcn_s_wait_tensorcnt)
#    define USE_TDM 1
#  else
#    define USE_TDM 0
#  endif
#else
#  define USE_TDM 0
#endif

static __device__ inline unsigned short f2bf(float f) {
    union { float f; unsigned int u; } v; v.f = f;
    unsigned int r = v.u + 0x7FFFu + ((v.u >> 16) & 1u);  // round-nearest-even
    return (unsigned short)(r >> 16);
}

static __device__ inline void unpack2(unsigned int u, __bf16& lo, __bf16& hi) {
    union { unsigned short s; __bf16 b; } a, b2;
    a.s  = (unsigned short)(u & 0xFFFFu);
    b2.s = (unsigned short)(u >> 16);
    lo = a.b; hi = b2.b;
}

// A fragment, 16x32 bf16: lanes 0-15 row M=lane hold K 0-15
// (dword-packed pairs), lanes 16-31 hold K 16-31.
static __device__ inline v16bf load_a_frag(const unsigned short* At, int m16, int lane) {
    int row  = m16 + (lane & 15);
    int half = (lane >> 4) & 1;
    const unsigned short* r = At + row * 32;
    v16bf a;
#pragma unroll
    for (int v = 0; v < 8; ++v) {
        int kk = half * 16 + (v << 1);
        unsigned int u = *(const unsigned int*)(r + kk);
        __bf16 lo, hi; unpack2(u, lo, hi);
        a[2 * v] = lo; a[2 * v + 1] = hi;
    }
    return a;
}

// B fragment, 32x16 bf16: lanes 0-15 col N=lane hold K 0-15, lanes 16-31 hold
// K 16-31.  Bt stored [n][k] so pairs are dword-contiguous.
static __device__ inline v16bf load_b_frag(const unsigned short* Bt, int n16, int lane) {
    int col   = n16 + (lane & 15);
    int kbase = (lane >> 4) << 4;
    const unsigned short* r = Bt + col * 32 + kbase;
    v16bf b;
#pragma unroll
    for (int v = 0; v < 8; ++v) {
        unsigned int u = *(const unsigned int*)(r + 2 * v);
        __bf16 lo, hi; unpack2(u, lo, hi);
        b[2 * v] = lo; b[2 * v + 1] = hi;
    }
    return b;
}

#if USE_TDM
typedef unsigned int u32x4 __attribute__((ext_vector_type(4)));
typedef int          i32x8 __attribute__((ext_vector_type(8)));
typedef int          i32x4 __attribute__((ext_vector_type(4)));

static __device__ inline unsigned lds_addr_of(const void* p) {
    return (unsigned)(unsigned long long)
           (__attribute__((address_space(3))) const void*)p;
}

// 2-D bf16 tile DMA: tile_w elements per row, tile_h rows, row stride in
// elements.  D# per CDNA5 ISA 8.3/8.4 (group0: count/lds/global/type;
// group1: data_size=2B, dims, tile dims, dim0 stride).  6-arg builtin on this
// toolchain; extra descriptor group zero-filled.
static __device__ inline void tdm_load_2d_bf16(unsigned lds_byte_addr,
                                               const unsigned short* gptr,
                                               unsigned tile_w, unsigned tile_h,
                                               unsigned row_stride) {
    unsigned long long ga = (unsigned long long)(const void*)gptr;
    u32x4 g0;
    g0[0] = 1u;                                         // count=1, user D#
    g0[1] = lds_byte_addr;                              // lds_addr
    g0[2] = (unsigned)(ga & 0xffffffffu);               // global_addr[31:0]
    g0[3] = (unsigned)((ga >> 32) & 0x1ffffffu)         // global_addr[56:32]
          | (2u << 30);                                 // type=2 (image)
    i32x8 g1;
    g1[0] = (int)(1u << 16);                            // data_size=1 -> 2B
    g1[1] = (int)((tile_w & 0xffffu) << 16);            // tensor_dim0[15:0]
    g1[2] = (int)(((tile_w >> 16) & 0xffffu)            // tensor_dim0[31:16]
          | ((tile_h & 0xffffu) << 16));                // tensor_dim1[15:0]
    g1[3] = (int)(((tile_h >> 16) & 0xffffu)
          | ((tile_w & 0xffffu) << 16));                // tile_dim0
    g1[4] = (int)(tile_h & 0xffffu);                    // tile_dim1, tile_dim2=0
    g1[5] = (int)row_stride;                            // tensor_dim0_stride lo
    g1[6] = 0;                                          // stride hi, dim1_stride
    g1[7] = 0;
    i32x4 z4; z4[0] = z4[1] = z4[2] = z4[3] = 0;        // 2-D: groups 2/3 unused
    i32x8 z8;
#pragma unroll
    for (int i = 0; i < 8; ++i) z8[i] = 0;
    __builtin_amdgcn_tensor_load_to_lds(g0, g1, z4, z4, z8, 0);
}
#endif

// ---------------------------------------------------------------------------
// 1) NCHW -> NHWC transpose, LDS tiled 32c x 64hw (65-pad: conflict-free).
__global__ __launch_bounds__(256) void k_transpose(const float* __restrict__ x,
                                                   float* __restrict__ xn) {
    __shared__ float t[32][65];
    int blk = blockIdx.x;                 // 2 * 4 * 256 = 2048 blocks
    int b   = blk >> 10;
    int r   = blk & 1023;
    int c0  = (r >> 8) * 32;
    int hw0 = (r & 255) * 64;
    int tid = threadIdx.x;
    int cl = tid >> 6, hl = tid & 63;
#pragma unroll
    for (int rr = 0; rr < 8; ++rr) {
        int c = cl + rr * 4;
        t[c][hl] = x[(size_t)(b * CC + c0 + c) * HW + hw0 + hl];
    }
    __syncthreads();
    int hl2 = tid >> 5, cl2 = tid & 31;
#pragma unroll
    for (int rr = 0; rr < 8; ++rr) {
        int hw = hl2 + rr * 8;
        xn[((size_t)b * HW + hw0 + hw) * CC + c0 + cl2] = t[cl2][hw];
    }
}

// 2a) repack offset+mod weights -> wA bf16 [32][1152], K = ij*128+c
__global__ __launch_bounds__(256) void k_prep_wA(const float* __restrict__ w_p,
                                                 const float* __restrict__ w_m,
                                                 unsigned short* __restrict__ wA) {
    unsigned q = blockIdx.x * 256u + threadIdx.x;       // 32*1152 = 36864
    int m = q / KTOT, k = q % KTOT;
    int ij = k >> 7,  c = k & 127;
    float v = 0.f;
    if (m < 18)      v = w_p[(m * CC + c) * 9 + ij];
    else if (m < 27) v = w_m[((m - 18) * CC + c) * 9 + ij];
    wA[q] = f2bf(v);
}

// 2b) repack final-conv weights -> wC bf16 [128][1152], K = ij*128+c
__global__ __launch_bounds__(256) void k_prep_wC(const float* __restrict__ w_conv,
                                                 unsigned short* __restrict__ wC) {
    unsigned q = blockIdx.x * 256u + threadIdx.x;       // 128*1152 = 147456
    int oc = q / KTOT, k = q % KTOT;
    int ij = k >> 7,   c = k & 127;
    wC[q] = f2bf(w_conv[(oc * CC + c) * 9 + ij]);
}

// 3) GEMM-A: offs[b][m][h*W+w] = sum_k wA[m][k] * im2col(x)[k][pix] + bias
//    One block per (b,h) row: M=32, N=128, K=1152, K-order ij*128+c so each
//    32-wide K-step has uniform (di,dj) shift.
__global__ __launch_bounds__(256) void k_gemmA(const float* __restrict__ x,
                                               const unsigned short* __restrict__ wA,
                                               const float* __restrict__ b_p,
                                               const float* __restrict__ b_m,
                                               float* __restrict__ offs) {
    __shared__ __align__(16) unsigned short At[32 * 32];
    __shared__ __align__(16) unsigned short Bt[128 * 32];
    int tid = threadIdx.x, wave = tid >> 5, lane = tid & 31;
    int blk = blockIdx.x;
    int b = blk >> 7, h = blk & 127;

    v8f acc[2];
#pragma unroll
    for (int t = 0; t < 2; ++t)
#pragma unroll
        for (int e = 0; e < 8; ++e) acc[t][e] = 0.f;

    for (int kt = 0; kt < KSTEPS; ++kt) {
        int k0 = kt * 32;
        int ij = k0 >> 7, c0 = k0 & 127;
        int di = ij / 3 - 1, dj = ij - (ij / 3) * 3 - 1;  // uniform per step
        int hh = h + di;
        if (tid < 128) {                                  // A tile 32x32, b128
            int m = tid >> 2, ch = tid & 3;
            *((uint4*)At + tid) = *(const uint4*)(wA + m * KTOT + k0 + ch * 8);
        }
        {   // B tile [n=w][k]: 16 channel-strided dword loads per thread,
            // one divergent region (ok is uniform across the 16 loads).
            int w  = tid >> 1, kh = (tid & 1) * 16;
            int ww = w + dj;
            bool ok = (hh >= 0) && (hh < HH) && (ww >= 0) && (ww < WW);
            unsigned int pk[8];
            if (ok) {
                const float* src =
                    x + (((size_t)b * CC + (c0 + kh)) * HH + hh) * WW + ww;
                float vv[16];
#pragma unroll
                for (int ee = 0; ee < 16; ++ee)
                    vv[ee] = src[(size_t)ee * HW];
#pragma unroll
                for (int ee = 0; ee < 8; ++ee)
                    pk[ee] = (unsigned)f2bf(vv[2 * ee])
                           | ((unsigned)f2bf(vv[2 * ee + 1]) << 16);
            } else {
#pragma unroll
                for (int ee = 0; ee < 8; ++ee) pk[ee] = 0u;
            }
            uint4 q0, q1;
            q0.x = pk[0]; q0.y = pk[1]; q0.z = pk[2]; q0.w = pk[3];
            q1.x = pk[4]; q1.y = pk[5]; q1.z = pk[6]; q1.w = pk[7];
            *(uint4*)(Bt + w * 32 + kh)     = q0;
            *(uint4*)(Bt + w * 32 + kh + 8) = q1;
        }
        __syncthreads();
        int m16 = (wave & 1) * 16;
        v16bf a = load_a_frag(At, m16, lane);
#pragma unroll
        for (int t = 0; t < 2; ++t) {
            int ns = (wave >> 1) * 2 + t;
            v16bf bb = load_b_frag(Bt, ns * 16, lane);
            acc[t] = __builtin_amdgcn_wmma_f32_16x16x32_bf16(
                false, a, false, bb, (short)0, acc[t], false, false);
        }
        __syncthreads();
    }

    int m16   = (wave & 1) * 16;
    int mbase = m16 + ((lane >> 4) << 3);
#pragma unroll
    for (int t = 0; t < 2; ++t) {
        int ns   = (wave >> 1) * 2 + t;
        int ncol = ns * 16 + (lane & 15);
#pragma unroll
        for (int r = 0; r < 8; ++r) {
            int M = mbase + r;
            if (M < 27) {
                float bias = (M < 18) ? b_p[M] : b_m[M - 18];
                offs[((size_t)b * 27 + M) * HW + h * WW + ncol] = acc[t][r] + bias;
            }
        }
    }
}

// 4a) zero G (padding taps of the final conv must be 0)
__global__ __launch_bounds__(256) void k_zero(uint4* __restrict__ p) {
    size_t i = (size_t)blockIdx.x * 256 + threadIdx.x;
    uint4 z; z.x = z.y = z.z = z.w = 0u;
    p[i] = z;
}

// 4b) deformable sampling: wave = one (b,pix), lane = 4 channels (float4).
//     Branch-free corners: zero-padding value 0 == bilinear weight 0, so
//     clamp indices into the interior and mask the weight instead.
__global__ __launch_bounds__(256) void k_sample(const float* __restrict__ xn,
                                                const float* __restrict__ offs,
                                                unsigned short* __restrict__ G) {
    int tid = threadIdx.x, wave = tid >> 5, lane = tid & 31;
    int gpix = blockIdx.x * 8 + wave;                   // 0..32767
    int b = gpix >> 14, p = gpix & 16383;
    int h = p >> 7, w = p & 127;
    const float4* xv = (const float4*)xn;

    for (int n = 0; n < 9; ++n) {
        int ki = n / 3, kj = n - ki * 3;
        float offx = offs[((size_t)b * 27 + n) * HW + p];
        float offy = offs[((size_t)b * 27 + 9 + n) * HW + p];
        float ml   = offs[((size_t)b * 27 + 18 + n) * HW + p];
        float m    = 1.f / (1.f + __expf(-ml));         // sigmoid

        float px = offx + (float)(ki - 1) + (float)(h + 1);
        float py = offy + (float)(kj - 1) + (float)(w + 1);
        float fx = floorf(px), fy = floorf(py);
        float ltx = fminf(fmaxf(fx, 0.f), 129.f);
        float lty = fminf(fmaxf(fy, 0.f), 129.f);
        float rbx = fminf(fmaxf(fx + 1.f, 0.f), 129.f);
        float rby = fminf(fmaxf(fy + 1.f, 0.f), 129.f);
        float cpx = fminf(fmaxf(px, 0.f), 129.f);
        float cpy = fminf(fmaxf(py, 0.f), 129.f);
        float glt = (1.f + (ltx - cpx)) * (1.f + (lty - cpy));
        float grb = (1.f - (rbx - cpx)) * (1.f - (rby - cpy));
        float glb = (1.f + (ltx - cpx)) * (1.f - (rby - cpy));
        float grt = (1.f - (rbx - cpx)) * (1.f + (lty - cpy));
        int ix0 = (int)ltx, iy0 = (int)lty, ix1 = (int)rbx, iy1 = (int)rby;

        // validity masks (padded coord in [1,128] hits real data, else 0)
        float vx0 = (ix0 >= 1 && ix0 <= HH) ? 1.f : 0.f;
        float vy0 = (iy0 >= 1 && iy0 <= WW) ? 1.f : 0.f;
        float vx1 = (ix1 >= 1 && ix1 <= HH) ? 1.f : 0.f;
        float vy1 = (iy1 >= 1 && iy1 <= WW) ? 1.f : 0.f;
        glt *= vx0 * vy0;
        grb *= vx1 * vy1;
        glb *= vx0 * vy1;
        grt *= vx1 * vy0;
        // clamped interior rows/cols (0..127)
        int cx0 = min(max(ix0, 1), HH) - 1, cy0 = min(max(iy0, 1), WW) - 1;
        int cx1 = min(max(ix1, 1), HH) - 1, cy1 = min(max(iy1, 1), WW) - 1;

        size_t rowb = (size_t)b * HH;
        float4 vlt = xv[((rowb + cx0) * WW + cy0) * 32 + lane];
        float4 vrb = xv[((rowb + cx1) * WW + cy1) * 32 + lane];
        float4 vlb = xv[((rowb + cx0) * WW + cy1) * 32 + lane];
        float4 vrt = xv[((rowb + cx1) * WW + cy0) * 32 + lane];

        float4 a;
        a.x = (glt * vlt.x + grb * vrb.x + glb * vlb.x + grt * vrt.x) * m;
        a.y = (glt * vlt.y + grb * vrb.y + glb * vlb.y + grt * vrt.y) * m;
        a.z = (glt * vlt.z + grb * vrb.z + glb * vlb.z + grt * vrt.z) * m;
        a.w = (glt * vlt.w + grb * vrb.w + glb * vlb.w + grt * vrt.w) * m;

        // Final conv tap that consumes this sample: i=(ki+1)%3, h_out=h+(ki==2)
        int io = (ki == 2) ? 0 : ki + 1;
        int jo = (kj == 2) ? 0 : kj + 1;
        int oh = h + (ki == 2 ? 1 : 0);
        int ow = w + (kj == 2 ? 1 : 0);
        if (oh < HH && ow < WW) {
            size_t base = (((size_t)(b * 9 + io * 3 + jo) * HW + oh * WW + ow) * CC)
                          + (size_t)lane * 4;
            unsigned int lo = (unsigned)f2bf(a.x) | ((unsigned)f2bf(a.y) << 16);
            unsigned int hi = (unsigned)f2bf(a.z) | ((unsigned)f2bf(a.w) << 16);
            uint2 pk; pk.x = lo; pk.y = hi;
            *(uint2*)(G + base) = pk;                   // 8B per lane, 256B/wave
        }
    }
}

// 5) GEMM-C: out[b][oc][pix] = sum_k wC[oc][k] * G[b][ij][pix][c], k=ij*128+c
//    Block tile 128(M) x 128(N), 8 waves, 8 WMMA/wave/K-step, TDM-staged
//    double-buffered tiles (wave 0 drives the DMA).
__global__ __launch_bounds__(256) void k_gemmC(const unsigned short* __restrict__ wC,
                                               const unsigned short* __restrict__ G,
                                               float* __restrict__ out) {
    __shared__ __align__(16) unsigned short At[2][128 * 32];
    __shared__ __align__(16) unsigned short Bt[2][128 * 32];
    int tid = threadIdx.x, wave = tid >> 5, lane = tid & 31;
    int blk = blockIdx.x;
    int b = blk >> 7, pt = blk & 127;
    int pixbase = pt * 128;

    v8f acc[8];
#pragma unroll
    for (int i = 0; i < 8; ++i)
#pragma unroll
        for (int e = 0; e < 8; ++e) acc[i][e] = 0.f;

#if USE_TDM
    // wave 0 drives the Tensor Data Mover; tiles are 2-D (32 K x 128 rows).
    if (wave == 0) {
        const unsigned short* a0 = wC;                               // k0 = 0
        const unsigned short* b0 = G + ((size_t)(b * 9) * HW + pixbase) * CC;
        tdm_load_2d_bf16(lds_addr_of(At[0]), a0, 32, 128, KTOT);
        tdm_load_2d_bf16(lds_addr_of(Bt[0]), b0, 32, 128, CC);
    }
    for (int kt = 0; kt < KSTEPS; ++kt) {
        int cur = kt & 1;
        if (wave == 0) __builtin_amdgcn_s_wait_tensorcnt(0);
        __syncthreads();                 // tiles[cur] ready for everyone
        if (wave == 0 && kt + 1 < KSTEPS) {
            int k0n = (kt + 1) * 32;
            int ijn = k0n >> 7, c0n = k0n & 127;
            const unsigned short* an = wC + k0n;
            const unsigned short* bn =
                G + ((size_t)(b * 9 + ijn) * HW + pixbase) * CC + c0n;
            tdm_load_2d_bf16(lds_addr_of(At[cur ^ 1]), an, 32, 128, KTOT);
            tdm_load_2d_bf16(lds_addr_of(Bt[cur ^ 1]), bn, 32, 128, CC);
        }
        v16bf a = load_a_frag(At[cur], wave * 16, lane);
#pragma unroll
        for (int ns = 0; ns < 8; ++ns) {
            v16bf bb = load_b_frag(Bt[cur], ns * 16, lane);
            acc[ns] = __builtin_amdgcn_wmma_f32_16x16x32_bf16(
                false, a, false, bb, (short)0, acc[ns], false, false);
        }
        __syncthreads();                 // done reading tiles[cur]
    }
#else
    for (int kt = 0; kt < KSTEPS; ++kt) {
        int k0 = kt * 32;
        int ij = k0 >> 7, c0 = k0 & 127;
        const unsigned short* gsrc =
            G + ((size_t)(b * 9 + ij) * HW + pixbase) * CC + c0;
#pragma unroll
        for (int it = 0; it < 2; ++it) {                // A tile, b128 copies
            int q = tid + it * 256;                     // 512 uint4 chunks
            int m = q >> 2, ch = q & 3;
            *((uint4*)At[0] + q) = *(const uint4*)(wC + m * KTOT + k0 + ch * 8);
        }
#pragma unroll
        for (int it = 0; it < 2; ++it) {                // B tile, b128 copies
            int q = tid + it * 256;
            int n = q >> 2, ch = q & 3;
            *((uint4*)Bt[0] + q) = *(const uint4*)(gsrc + (size_t)n * CC + ch * 8);
        }
        if (kt + 1 < KSTEPS) {
            int k0n = (kt + 1) * 32;
            int ijn = k0n >> 7, c0n = k0n & 127;
            __builtin_prefetch(G + ((size_t)(b * 9 + ijn) * HW + pixbase
                                    + (tid >> 1)) * CC + c0n, 0, 0);
        }
        __syncthreads();
        v16bf a = load_a_frag(At[0], wave * 16, lane);
#pragma unroll
        for (int ns = 0; ns < 8; ++ns) {
            v16bf bb = load_b_frag(Bt[0], ns * 16, lane);
            acc[ns] = __builtin_amdgcn_wmma_f32_16x16x32_bf16(
                false, a, false, bb, (short)0, acc[ns], false, false);
        }
        __syncthreads();
    }
#endif

    int mbase = wave * 16 + ((lane >> 4) << 3);
#pragma unroll
    for (int ns = 0; ns < 8; ++ns) {
        int ncol = pixbase + ns * 16 + (lane & 15);
#pragma unroll
        for (int r = 0; r < 8; ++r) {
            int M = mbase + r;
            out[((size_t)(b * 128 + M)) * HW + ncol] = acc[ns][r];
        }
    }
}

// ---------------------------------------------------------------------------
extern "C" void kernel_launch(void* const* d_in, const int* in_sizes, int n_in,
                              void* d_out, int out_size, void* d_ws, size_t ws_size,
                              hipStream_t stream) {
    const float* x      = (const float*)d_in[0];
    const float* w_p    = (const float*)d_in[1];
    const float* b_p    = (const float*)d_in[2];
    const float* w_m    = (const float*)d_in[3];
    const float* b_m    = (const float*)d_in[4];
    const float* w_conv = (const float*)d_in[5];
    float* out = (float*)d_out;

    char* ws = (char*)d_ws;
    float*          xn   = (float*)(ws + 0);                   // 16,777,216 B
    float*          offs = (float*)(ws + 16777216);            //  3,538,944 B
    unsigned short* wA   = (unsigned short*)(ws + 20316160);   //     73,728 B
    unsigned short* wC   = (unsigned short*)(ws + 20389888);   //    294,912 B
    unsigned short* G    = (unsigned short*)(ws + 20684800);   // 75,497,472 B

    k_transpose<<<2048,  256, 0, stream>>>(x, xn);
    k_prep_wA  <<<144,   256, 0, stream>>>(w_p, w_m, wA);
    k_prep_wC  <<<576,   256, 0, stream>>>(w_conv, wC);
    k_gemmA    <<<256,   256, 0, stream>>>(x, wA, b_p, b_m, offs);
    k_zero     <<<18432, 256, 0, stream>>>((uint4*)G);         // 75,497,472/16
    k_sample   <<<4096,  256, 0, stream>>>(xn, offs, G);
    k_gemmC    <<<256,   256, 0, stream>>>(wC, G, out);
}
